// SAGEConv_26723286516151
// MI455X (gfx1250) — matrix operand order repaired and checked by reading, not compiled, not run on hardware
//
#include <hip/hip_runtime.h>

typedef _Float16 half8_t __attribute__((ext_vector_type(8)));
typedef _Float16 v16h    __attribute__((ext_vector_type(16)));
typedef float    v8f     __attribute__((ext_vector_type(8)));

#define N_NODES 20000
#define DEG     32
#define D_IN    128
#define D_OUT   256
#define K_TOT   256   // fused K: [x | neigh]

// ---------------------------------------------------------------------------
// Kernel 1: weighted-mean aggregation + f16 pack of A' = [x | neigh], [N, 256]
// One block (128 threads) per node; edge list is contiguous per node.
// ---------------------------------------------------------------------------
__global__ void __launch_bounds__(128)
sage_agg_pack(const float* __restrict__ x,
              const int*   __restrict__ col,
              const float* __restrict__ ew,
              _Float16*    __restrict__ Ah) {
  __shared__ float sw[DEG];
  __shared__ int   sc[DEG];
  const int node = blockIdx.x;
  const int t    = threadIdx.x;          // feature 0..127
  if (t < DEG) {
    sw[t] = ew[node * DEG + t];
    sc[t] = col[node * DEG + t];
  }
  __syncthreads();

  float acc = 0.0f, deg = 0.0f;
#pragma unroll 8
  for (int e = 0; e < DEG; ++e) {
    const float w = sw[e];
    acc += w * x[(size_t)sc[e] * D_IN + t];
    deg += w;
  }
  if (deg == 0.0f) deg = 1.0f;

  Ah[(size_t)node * K_TOT + t]        = (_Float16)x[(size_t)node * D_IN + t];
  Ah[(size_t)node * K_TOT + D_IN + t] = (_Float16)(acc / deg);
}

// ---------------------------------------------------------------------------
// Kernel 2: pack B' = [W_l ; W_r] (256x256 f32, row-major K x Nout) into the
// per-lane WMMA B-operand order for v_wmma_f32_16x16x32_f16:
//   element p = (((tile_n*8 + s)*32 + lane)*16 + j)
//   maps to  K = 32*s + 8*(lane/16) + (j%8) + 16*(j/8),  N = 16*tile_n + (lane%16)
// so the GEMM reads each lane's 16 halves as one contiguous 32B chunk.
// ---------------------------------------------------------------------------
__global__ void __launch_bounds__(256)
sage_pack_b(const float* __restrict__ Wl,
            const float* __restrict__ Wr,
            _Float16*    __restrict__ Bp) {
  const int p    = blockIdx.x * blockDim.x + threadIdx.x;  // 0 .. 65535
  const int j    = p & 15;
  const int lane = (p >> 4) & 31;
  const int s    = (p >> 9) & 7;
  const int tn   = p >> 12;
  const int k    = s * 32 + ((lane >> 4) << 3) + (j & 7) + ((j >> 3) << 4);
  const int n    = tn * 16 + (lane & 15);
  const float v  = (k < D_IN) ? Wl[k * D_OUT + n] : Wr[(k - D_IN) * D_OUT + n];
  Bp[p] = (_Float16)v;
}

// ---------------------------------------------------------------------------
// Kernel 3: out[M, 256] = A'[M, 256] @ B'[256, 256] + b_l, via WMMA.
// One wave per 16x16 tile; K=256 -> 8 x v_wmma_f32_16x16x32_f16 (f32 accum).
// A-operand (16-bit A 16x32, ISA 7.12.2): lane row = lane%16, holds K-chunks
// [8*(lane/16) .. +7] and the same +16 -> two aligned 16B loads per k-step.
// C/D f32 layout: VGPR r -> row (r + 8*(lane/16)), col (lane%16).
// ---------------------------------------------------------------------------
__global__ void __launch_bounds__(256)
sage_wmma_gemm(const _Float16* __restrict__ Ah,
               const _Float16* __restrict__ Bp,
               const float*    __restrict__ bias,
               float*          __restrict__ out) {
  const int gwid   = (int)((blockIdx.x * blockDim.x + threadIdx.x) >> 5);
  const int lane   = threadIdx.x & 31;
  const int tile_m = gwid >> 4;           // 0..1249  (wave-uniform)
  const int tile_n = gwid & 15;           // 0..15    (wave-uniform)
  if (tile_m >= N_NODES / 16) return;     // uniform; EXEC stays all-1s

  const int m0    = tile_m << 4;
  const int n0    = tile_n << 4;
  const int lhalf = lane >> 4;            // 0 or 1
  const int lmod  = lane & 15;

  const _Float16* pa = Ah + (size_t)(m0 + lmod) * K_TOT + (lhalf << 3);
  const _Float16* pb = Bp + ((size_t)(tile_n * 8) * 32 + lane) * 16;

  v8f c = {0.f, 0.f, 0.f, 0.f, 0.f, 0.f, 0.f, 0.f};

#pragma unroll
  for (int s = 0; s < 8; ++s) {
    union { v16h v; half8_t h[2]; } a, b;
    a.h[0] = *(const half8_t*)(pa + s * 32);        // K = 32s + 8*lhalf .. +7
    a.h[1] = *(const half8_t*)(pa + s * 32 + 16);   // K = +16
    b.h[0] = *(const half8_t*)(pb + (size_t)s * 512);       // lane's 16 halves,
    b.h[1] = *(const half8_t*)(pb + (size_t)s * 512 + 8);   // contiguous 32B
    // (neg_a, A, neg_b, B, c_mod, C, reuse_a, reuse_b)
    c = __builtin_amdgcn_wmma_f32_16x16x32_f16(
        false, a.v, false, b.v, (short)0, c, false, false);
  }

  const float bn = bias[n0 + lmod];
  float* po = out + (size_t)(m0 + (lhalf << 3)) * D_OUT + n0 + lmod;
#pragma unroll
  for (int r = 0; r < 8; ++r) {
    po[(size_t)r * D_OUT] = c[r] + bn;   // row m0 + r + 8*lhalf, col n0 + lmod
  }
}

// ---------------------------------------------------------------------------
extern "C" void kernel_launch(void* const* d_in, const int* in_sizes, int n_in,
                              void* d_out, int out_size, void* d_ws, size_t ws_size,
                              hipStream_t stream) {
  (void)in_sizes; (void)n_in; (void)out_size; (void)ws_size;

  const float* x  = (const float*)d_in[0];              // [N, 128]
  const int*   ei = (const int*)  d_in[1];              // [2, E]
  const float* ew = (const float*)d_in[2];              // [E]
  const float* Wl = (const float*)d_in[3];              // [128, 256]
  const float* bl = (const float*)d_in[4];              // [256]
  const float* Wr = (const float*)d_in[5];              // [128, 256]
  const int*  col = ei + (size_t)N_NODES * DEG;         // edge_index[1]

  _Float16* Ah = (_Float16*)d_ws;                                        // 10.24 MB
  _Float16* Bp = (_Float16*)((char*)d_ws +
                             (size_t)N_NODES * K_TOT * sizeof(_Float16)); // 128 KB
  float* out = (float*)d_out;

  sage_agg_pack<<<N_NODES, 128, 0, stream>>>(x, col, ew, Ah);
  sage_pack_b<<<(K_TOT * D_OUT) / 256, 256, 0, stream>>>(Wl, Wr, Bp);

  const int total_waves = (N_NODES / 16) * (D_OUT / 16);   // 20000 tiles
  const int blocks      = total_waves / 8;                 // 8 waves / 256-thr block
  sage_wmma_gemm<<<blocks, 256, 0, stream>>>(Ah, Bp, bl, out);
}
